// TreeLSTM_81243601371885
// MI455X (gfx1250) — matrix operand order repaired and checked by reading, not compiled, not run on hardware
//
#include <hip/hip_runtime.h>
#include <stdint.h>

#define N_TREES 512
#define DEPTH   10
#define NPT     1023                 // nodes per tree
#define NTOT    (N_TREES * NPT)      // 523776
#define HSZ     128

typedef __attribute__((ext_vector_type(16))) __bf16   v16bf;
typedef __attribute__((ext_vector_type(8)))  float    v8f;
typedef __attribute__((ext_vector_type(4)))  float    f32x4;
typedef __attribute__((ext_vector_type(4)))  unsigned u32x4;
typedef __attribute__((ext_vector_type(2)))  unsigned u32x2;

union Frag { v16bf v; u32x4 q[2]; };

__device__ __forceinline__ unsigned short f2bf(float f) {
    unsigned u = __float_as_uint(f);
    u += 0x7fffu + ((u >> 16) & 1u);           // round-to-nearest-even
    return (unsigned short)(u >> 16);
}
__device__ __forceinline__ unsigned pk(float a, float b) {
    return (unsigned)f2bf(a) | ((unsigned)f2bf(b) << 16);
}
__device__ __forceinline__ float sigm(float x) { return 1.0f / (1.0f + __expf(-x)); }

// A fragment (16x32 bf16, row-major tile in LDS, row stride 136 ushorts):
// lane: m = lane%16, half = lane/16; elems 0..15 -> K = k0+half*16+e
__device__ __forceinline__ void loadA(Frag& f, const unsigned short* tile, int m, int k0, int half) {
    const u32x4* p = (const u32x4*)(tile + m * 136 + k0 + half * 16);
    f.q[0] = p[0];
    f.q[1] = p[1];
}
// B fragment (32x16 bf16): lane: n = lane%16, half = lane/16; elems 0..15 -> K = k0+half*16+e
// B[k][n] = W[row_n][k] (W row-major [*,128]) -> 32 contiguous bytes per lane.
__device__ __forceinline__ void loadB(Frag& f, const unsigned short* W, int row, int k0, int half) {
    const u32x4* p = (const u32x4*)(W + row * 128 + k0 + half * 16);
    f.q[0] = p[0];
    f.q[1] = p[1];
}
__device__ __forceinline__ v8f wmma_bf16(const Frag& a, const Frag& b, v8f c) {
    return __builtin_amdgcn_wmma_f32_16x16x32_bf16(false, a.v, false, b.v, (short)0, c, false, false);
}
__device__ __forceinline__ v8f splat8(float x) {
    v8f r = {x, x, x, x, x, x, x, x};
    return r;
}

// ---------------- prep: weights -> bf16, fuse biases ----------------
__global__ void prep_kernel(const float* __restrict__ W_iou, const float* __restrict__ U_iou,
                            const float* __restrict__ W_f,   const float* __restrict__ U_f,
                            const float* __restrict__ b_iou, const float* __restrict__ b_Uiou,
                            const float* __restrict__ b_Wf,  const float* __restrict__ b_Uf,
                            unsigned short* __restrict__ Wi, unsigned short* __restrict__ Ui,
                            unsigned short* __restrict__ Wf, unsigned short* __restrict__ Uf,
                            float* __restrict__ Bi, float* __restrict__ Bf) {
    int t = blockIdx.x * 256 + threadIdx.x;
    if (t < 49152) { Wi[t] = f2bf(W_iou[t]); Ui[t] = f2bf(U_iou[t]); }
    if (t < 16384) { Wf[t] = f2bf(W_f[t]);   Uf[t] = f2bf(U_f[t]); }
    if (t < 384)   { Bi[t] = b_iou[t] + b_Uiou[t]; }
    if (t < 128)   { Bf[t] = b_Wf[t] + b_Uf[t]; }
}

// ---------------- per-level TreeLSTM cell, 32 nodes per wave (2 M-tiles) ----------------
template <bool LEAF>
__global__ __launch_bounds__(64) void level_kernel(
    const int* __restrict__ wordid, const float* __restrict__ E,
    const unsigned short* __restrict__ Wi, const unsigned short* __restrict__ Ui,
    const unsigned short* __restrict__ Wf, const unsigned short* __restrict__ Uf,
    const float* __restrict__ Bi, const float* __restrict__ Bf,
    float* __restrict__ h, float* __restrict__ c,
    int start, int lcnt) {
    __shared__ unsigned short sX [2][32][136];
    __shared__ unsigned short sHT[2][32][136];
    __shared__ unsigned short sH0[2][32][136];
    __shared__ unsigned short sH1[2][32][136];

    const int lane = threadIdx.x & 31;
    const int wave = threadIdx.x >> 5;
    const int tileBase = (blockIdx.x * 2 + wave) * 32;   // 32 nodes per wave
    const int col4 = lane * 4;

    // ---- stage x / h_children / h_tild as bf16 tiles in LDS (32 rows) ----
    for (int r = 0; r < 32; ++r) {
        int idx  = tileBase + r;
        int tree = idx >> lcnt;
        int heap = start + (idx - (tree << lcnt));
        int nid  = tree * NPT + heap;
        int wid  = wordid[nid];
        f32x4 xv = (f32x4)0.0f;
        if (wid >= 0) xv = *(const f32x4*)(E + (size_t)wid * HSZ + col4);
        *(u32x2*)&sX[wave][r][col4] = (u32x2){ pk(xv[0], xv[1]), pk(xv[2], xv[3]) };
        if (!LEAF) {
            int cb = tree * NPT + 2 * heap;
            f32x4 h0 = *(const f32x4*)(h + (size_t)(cb + 1) * HSZ + col4);
            f32x4 h1 = *(const f32x4*)(h + (size_t)(cb + 2) * HSZ + col4);
            f32x4 ht = h0 + h1;
            *(u32x2*)&sH0[wave][r][col4] = (u32x2){ pk(h0[0], h0[1]), pk(h0[2], h0[3]) };
            *(u32x2*)&sH1[wave][r][col4] = (u32x2){ pk(h1[0], h1[1]), pk(h1[2], h1[3]) };
            *(u32x2*)&sHT[wave][r][col4] = (u32x2){ pk(ht[0], ht[1]), pk(ht[2], ht[3]) };
        }
    }
    __syncthreads();

    const int n    = lane & 15;   // A-row within tile, also C/D column
    const int half = lane >> 4;

    const unsigned short* ldsX  = &sX [wave][0][0];
    const unsigned short* ldsHT = &sHT[wave][0][0];
    const unsigned short* ldsH0 = &sH0[wave][0][0];
    const unsigned short* ldsH1 = &sH1[wave][0][0];
    const int HIT = 16 * 136;     // offset of high M-tile rows in a staged buffer

    for (int jt = 0; jt < 8; ++jt) {
        const int row = jt * 16 + n;              // weight-matrix output row
        const float bi_i = Bi[row], bi_o = Bi[128 + row], bi_u = Bi[256 + row];
        const float bf_v = LEAF ? 0.0f : Bf[row];
        v8f iL = splat8(bi_i), iH = splat8(bi_i);
        v8f oL = splat8(bi_o), oH = splat8(bi_o);
        v8f uL = splat8(bi_u), uH = splat8(bi_u);
        v8f fxL = splat8(bf_v), fxH = splat8(bf_v);

        // ---- K segment 1: x vs W_iou / W_f (W_f*x shared by both children) ----
#pragma unroll
        for (int ks = 0; ks < 4; ++ks) {
            const int k0 = ks * 32;
            Frag b_i, b_o, b_u, b_f, aL, aH;
            loadB(b_i, Wi, row,       k0, half);
            loadB(b_o, Wi, 128 + row, k0, half);
            loadB(b_u, Wi, 256 + row, k0, half);
            if (!LEAF) loadB(b_f, Wf, row, k0, half);
            loadA(aL, ldsX,       n, k0, half);
            loadA(aH, ldsX + HIT, n, k0, half);
            iL = wmma_bf16(aL, b_i, iL);  iH = wmma_bf16(aH, b_i, iH);
            oL = wmma_bf16(aL, b_o, oL);  oH = wmma_bf16(aH, b_o, oH);
            uL = wmma_bf16(aL, b_u, uL);  uH = wmma_bf16(aH, b_u, uH);
            if (!LEAF) { fxL = wmma_bf16(aL, b_f, fxL); fxH = wmma_bf16(aH, b_f, fxH); }
        }
        v8f f0L = fxL, f0H = fxH;   // fork shared W_f*x + b into both children
        v8f f1L = fxL, f1H = fxH;
        if (!LEAF) {
            // ---- K segment 2: h_tild vs U_iou; h_c0/h_c1 vs U_f (B shared) ----
#pragma unroll
            for (int ks = 0; ks < 4; ++ks) {
                const int k0 = ks * 32;
                Frag b_i, b_o, b_u, b_f, aL, aH;
                loadB(b_i, Ui, row,       k0, half);
                loadB(b_o, Ui, 128 + row, k0, half);
                loadB(b_u, Ui, 256 + row, k0, half);
                loadB(b_f, Uf, row,       k0, half);
                loadA(aL, ldsHT,       n, k0, half);
                loadA(aH, ldsHT + HIT, n, k0, half);
                iL = wmma_bf16(aL, b_i, iL);  iH = wmma_bf16(aH, b_i, iH);
                oL = wmma_bf16(aL, b_o, oL);  oH = wmma_bf16(aH, b_o, oH);
                uL = wmma_bf16(aL, b_u, uL);  uH = wmma_bf16(aH, b_u, uH);
                loadA(aL, ldsH0,       n, k0, half);
                loadA(aH, ldsH0 + HIT, n, k0, half);
                f0L = wmma_bf16(aL, b_f, f0L); f0H = wmma_bf16(aH, b_f, f0H);
                loadA(aL, ldsH1,       n, k0, half);
                loadA(aH, ldsH1 + HIT, n, k0, half);
                f1L = wmma_bf16(aL, b_f, f1L); f1H = wmma_bf16(aH, b_f, f1H);
            }
        }
        // ---- gates + state update (two M-tiles) ----
        const int j = jt * 16 + n;
#pragma unroll
        for (int t = 0; t < 2; ++t) {
            const v8f& ai  = t ? iH : iL;
            const v8f& ao  = t ? oH : oL;
            const v8f& au  = t ? uH : uL;
            const v8f& af0 = t ? f0H : f0L;
            const v8f& af1 = t ? f1H : f1L;
#pragma unroll
            for (int r = 0; r < 8; ++r) {
                int m    = t * 16 + r + half * 8;        // row within 32-node tile
                int idx  = tileBase + m;
                int tree = idx >> lcnt;
                int heap = start + (idx - (tree << lcnt));
                int nOff = (tree * NPT + heap) * HSZ;
                float iv = sigm(ai[r]);
                float ov = sigm(ao[r]);
                float uv = tanhf(au[r]);
                float ct = iv * uv;
                if (!LEAF) {
                    int c0Off = (tree * NPT + 2 * heap + 1) * HSZ;
                    float f0 = sigm(af0[r]);
                    float f1 = sigm(af1[r]);
                    ct += f0 * c[c0Off + j] + f1 * c[c0Off + HSZ + j];
                }
                float hv = ov * tanhf(ct);
                c[nOff + j] = ct;
                h[nOff + j] = hv;
            }
        }
    }
}

// ---------------- logits = h @ W_out^T + b_out ----------------
__global__ __launch_bounds__(256) void logits_kernel(const float* __restrict__ h,
                                                     const float* __restrict__ W_out,
                                                     const float* __restrict__ b_out,
                                                     float* __restrict__ out) {
    __shared__ float sW[5 * 128];
    __shared__ float sb[5];
    for (int t = threadIdx.x; t < 640; t += 256) sW[t] = W_out[t];
    if (threadIdx.x < 5) sb[threadIdx.x] = b_out[threadIdx.x];
    __syncthreads();
    int node = blockIdx.x * 256 + threadIdx.x;           // grid = NTOT/256 exactly
    const float* hr = h + (size_t)node * HSZ;
    float a0 = sb[0], a1 = sb[1], a2 = sb[2], a3 = sb[3], a4 = sb[4];
    for (int k = 0; k < 128; k += 4) {
        f32x4 hv = *(const f32x4*)(hr + k);
#pragma unroll
        for (int q = 0; q < 4; ++q) {
            float v = hv[q];
            a0 += v * sW[0 * 128 + k + q];
            a1 += v * sW[1 * 128 + k + q];
            a2 += v * sW[2 * 128 + k + q];
            a3 += v * sW[3 * 128 + k + q];
            a4 += v * sW[4 * 128 + k + q];
        }
    }
    float* o = out + (size_t)node * 5;
    o[0] = a0; o[1] = a1; o[2] = a2; o[3] = a3; o[4] = a4;
}

extern "C" void kernel_launch(void* const* d_in, const int* in_sizes, int n_in,
                              void* d_out, int out_size, void* d_ws, size_t ws_size,
                              hipStream_t stream) {
    (void)in_sizes; (void)n_in; (void)out_size; (void)ws_size;
    const int*   wordid = (const int*)  d_in[0];
    const float* E      = (const float*)d_in[1];
    const float* W_iou  = (const float*)d_in[2];
    const float* b_iou  = (const float*)d_in[3];
    const float* U_iou  = (const float*)d_in[4];
    const float* b_Uiou = (const float*)d_in[5];
    const float* W_f    = (const float*)d_in[6];
    const float* b_Wf   = (const float*)d_in[7];
    const float* U_f    = (const float*)d_in[8];
    const float* b_Uf   = (const float*)d_in[9];
    const float* W_out  = (const float*)d_in[10];
    const float* b_out  = (const float*)d_in[11];

    char* ws = (char*)d_ws;
    float*          h  = (float*)(ws + 0ull);
    float*          c  = (float*)(ws + 268173312ull);       // NTOT*128*4
    unsigned short* Wi = (unsigned short*)(ws + 536346624ull);
    unsigned short* Ui = (unsigned short*)(ws + 536444928ull);
    unsigned short* Wf = (unsigned short*)(ws + 536543232ull);
    unsigned short* Uf = (unsigned short*)(ws + 536576000ull);
    float*          Bi = (float*)(ws + 536608768ull);
    float*          Bf = (float*)(ws + 536610304ull);

    prep_kernel<<<192, 256, 0, stream>>>(W_iou, U_iou, W_f, U_f,
                                         b_iou, b_Uiou, b_Wf, b_Uf,
                                         Wi, Ui, Wf, Uf, Bi, Bf);

    for (int l = DEPTH - 1; l >= 0; --l) {
        int start = (1 << l) - 1;
        int M = N_TREES << l;                 // nodes at this level
        dim3 grid(M / 64);                    // 2 waves x 32 nodes per block
        if (l == DEPTH - 1)
            level_kernel<true ><<<grid, 64, 0, stream>>>(wordid, E, Wi, Ui, Wf, Uf, Bi, Bf, h, c, start, l);
        else
            level_kernel<false><<<grid, 64, 0, stream>>>(wordid, E, Wi, Ui, Wf, Uf, Bi, Bf, h, c, start, l);
    }

    logits_kernel<<<NTOT / 256, 256, 0, stream>>>(h, W_out, b_out, (float*)d_out);
}
